// Imputer_40724879900971
// MI455X (gfx1250) — compile-verified, matrix operand order checked by reading, not run on hardware
//
#include <hip/hip_runtime.h>
#include <hip/hip_bf16.h>

// Problem constants (from reference)
#define B_  32
#define V_  512
#define Q_  100
#define K_  5
#define E_  25
#define DM_ 30
#define H_  6
#define L_  6
#define DF_ 512
#define DH_ 5
#define DMP 32          // DM padded to 32 for WMMA K/N (row30=0, row31=bias row)
#define EPS_ 1e-6f

typedef __attribute__((ext_vector_type(16))) _Float16 v16h;
typedef __attribute__((ext_vector_type(8)))  _Float16 v8h;
typedef __attribute__((ext_vector_type(8)))  float    v8f;
typedef __attribute__((ext_vector_type(4)))  unsigned int v4u;
typedef __attribute__((ext_vector_type(8)))  int      v8i;
typedef __attribute__((ext_vector_type(4)))  int      v4i;

// ---------------- LDS layout (bytes) ----------------
// xs  : f32 [512][32]  x / prev                 (64 KB)
// hs  : f32 [512][32]  h (post-attn, final h)   (64 KB)  (start: TDM landing for inputs)
// hn  : f16 [512][32]  LN output (WMMA A)       (32 KB)
// qo  : f16 [512][32]  q, then o (token-major)  (32 KB)
// kh  : f16 [6][512][8] head-major K (pad 8)    (48 KB)  \ aliased post-attention by
// vh  : f16 [6][512][8] head-major V            (48 KB)  / W1F(32K)+W2F(32K)+US(32K)
// wq4 : f16 QKVO B-fragments [4][2][32][16]     ( 8 KB)
#define OFF_XS 0
#define OFF_HS (OFF_XS + V_*DMP*4)
#define OFF_HN (OFF_HS + V_*DMP*4)
#define OFF_QO (OFF_HN + V_*DMP*2)
#define OFF_KH (OFF_QO + V_*DMP*2)
#define OFF_VH (OFF_KH + H_*V_*8*2)
#define OFF_W1F (OFF_KH)                 // 32 KB (alias kh)
#define OFF_W2F (OFF_W1F + 32*32*16*2)   // 32 KB (alias kh/vh)
#define OFF_US  (OFF_W2F + 32*32*16*2)   // 32 KB (alias vh)
#define OFF_WQ4 (OFF_VH + H_*V_*8*2)
#define SMEM_BYTES (OFF_WQ4 + 4*2*32*16*2)   // 303104 B < 320 KB WGP LDS

#if defined(__has_builtin)
#if __has_builtin(__builtin_amdgcn_tensor_load_to_lds) && __has_builtin(__builtin_amdgcn_s_wait_tensorcnt)
#define HAVE_TDM 1
#endif
#endif

// B-fragment element (i in 0..15) <-> (k,n) of a 32x16 tile, per ISA mirrored layout
__device__ __forceinline__ int frag_k(int lane, int i) {
  const int kb = (lane >> 4) * 8;
  return (i < 8) ? (kb + i) : (kb + 16 + (i - 8));
}

// ---- A-fragment (16x32 f16) from LDS row-major: two aligned b128 loads ----
__device__ __forceinline__ v16h load_afrag(const _Float16* base, int ld) {
  const int lane = threadIdx.x & 31;
  const int m  = lane & 15;
  const int kb = (lane >> 4) * 8;
  const _Float16* p = base + m * ld + kb;
  const v8h lo = *(const v8h*)p;
  const v8h hi = *(const v8h*)(p + 16);
  return __builtin_shufflevector(lo, hi, 0,1,2,3,4,5,6,7,8,9,10,11,12,13,14,15);
}

// ---- B-fragment from LDS fragment-order stage: [frag][lane][16] f16 ----
__device__ __forceinline__ v16h load_frag16(const _Float16* base, int frag) {
  const int lane = threadIdx.x & 31;
  const _Float16* p = base + ((frag * 32 + lane) << 4);
  const v8h lo = *(const v8h*)p;
  const v8h hi = *(const v8h*)(p + 8);
  return __builtin_shufflevector(lo, hi, 0,1,2,3,4,5,6,7,8,9,10,11,12,13,14,15);
}

__device__ __forceinline__ v8f wmma16(v16h a, v16h b, v8f c) {
  return __builtin_amdgcn_wmma_f32_16x16x32_f16(false, a, false, b,
                                                (short)0, c, false, false);
}

__global__ __launch_bounds__(1024, 1)
void imputer_fused_kernel(const float* __restrict__ inputs,
                          const int*   __restrict__ questions,
                          const float* __restrict__ q_emb,
                          const float* __restrict__ q_alpha,
                          const float* __restrict__ q_bias,
                          const float* __restrict__ Wq, const float* __restrict__ bq,
                          const float* __restrict__ Wk, const float* __restrict__ bk,
                          const float* __restrict__ Wv, const float* __restrict__ bv,
                          const float* __restrict__ Wo, const float* __restrict__ bo,
                          const float* __restrict__ n1a, const float* __restrict__ n1b,
                          const float* __restrict__ n2a, const float* __restrict__ n2b,
                          const float* __restrict__ W1, const float* __restrict__ b1,
                          const float* __restrict__ W2, const float* __restrict__ b2,
                          float* __restrict__ out) {
  extern __shared__ char smem[];
  float*    xs  = (float*)(smem + OFF_XS);
  float*    hs  = (float*)(smem + OFF_HS);
  _Float16* hn  = (_Float16*)(smem + OFF_HN);
  _Float16* qo  = (_Float16*)(smem + OFF_QO);
  _Float16* kh  = (_Float16*)(smem + OFF_KH);
  _Float16* vh  = (_Float16*)(smem + OFF_VH);
  _Float16* w1f = (_Float16*)(smem + OFF_W1F);
  _Float16* w2f = (_Float16*)(smem + OFF_W2F);
  _Float16* uS  = (_Float16*)(smem + OFF_US);
  _Float16* wq4 = (_Float16*)(smem + OFF_WQ4);

  const int b    = blockIdx.x;
  const int tid  = threadIdx.x;
  const int lane = tid & 31;
  const int w    = tid >> 5;     // wave id 0..31
  const int m0   = w * 16;       // this wave's M-tile rows

  const int qi = (tid < V_) ? questions[b * V_ + tid] : 0;

  // ---------------- TDM: DMA this block's inputs slice (512x6 f32) into hs ----------------
#ifdef HAVE_TDM
  if (w == 0) {
    const unsigned long long ga =
        (unsigned long long)(const void*)(inputs + (size_t)b * V_ * (K_ + 1));
    v4u g0 = {};
    g0[0] = 1u;                                        // count=1 (valid user descriptor)
    g0[1] = (unsigned)OFF_HS;                          // lds_addr (bytes)
    g0[2] = (unsigned)(ga & 0xFFFFFFFFu);              // global_addr[31:0]
    g0[3] = (unsigned)((ga >> 32) & 0x1FFFFFFu) | (2u << 30);  // addr[56:32] | type=2
    v8i g1 = {};
    g1[0] = (2 << 16);                                 // data_size = 4 bytes
    g1[1] = (int)((unsigned)(V_ * (K_ + 1) & 0xFFFF) << 16);   // tensor_dim0 lo16
    g1[2] = 0;                                         // tensor_dim0 hi16 / dim1 lo16
    g1[3] = (int)((unsigned)(V_ * (K_ + 1) & 0xFFFF) << 16);   // tile_dim0 (1D tile)
    g1[4] = 0;                                         // tile_dim1/2 unused
    g1[5] = V_ * (K_ + 1);                             // tensor_dim0_stride lo32
    g1[6] = 0; g1[7] = 0;
    v4i gz = {};
#if __has_include(<hip/amd_detail/amd_gfx1250_TDM.h>)
    v8i g4 = {};
    __builtin_amdgcn_tensor_load_to_lds(g0, g1, gz, gz, g4, 0);
#else
    __builtin_amdgcn_tensor_load_to_lds(g0, g1, gz, gz, 0);
#endif
    __builtin_amdgcn_s_wait_tensorcnt(0);
  }
  __syncthreads();
#endif

  // ---------------- build initial x ----------------
  if (tid < V_) {
#ifdef HAVE_TDM
    const float* inp = hs + tid * (K_ + 1);
#else
    const float* inp = inputs + ((size_t)b * V_ + tid) * (K_ + 1);
#endif
    const bool obs = (inp[0] == 0.0f);
    float* xrow = xs + tid * DMP;
#pragma unroll
    for (int e = 0; e < E_; ++e) xrow[e] = q_emb[qi * E_ + e];
#pragma unroll
    for (int k = 0; k < K_; ++k) xrow[E_ + k] = obs ? inp[1 + k] : (1.0f / (float)K_);
    xrow[30] = 0.0f; xrow[31] = 0.0f;
  }
  __syncthreads();

  for (int l = 0; l < L_; ++l) {
    const float* Wq_l = Wq + (size_t)l * DM_ * DM_;
    const float* Wk_l = Wk + (size_t)l * DM_ * DM_;
    const float* Wv_l = Wv + (size_t)l * DM_ * DM_;
    const float* Wo_l = Wo + (size_t)l * DM_ * DM_;
    const float* W1_l = W1 + (size_t)l * DM_ * DF_;
    const float* W2_l = W2 + (size_t)l * DF_ * DM_;
    const float* b1_l = b1 + l * DF_;
    const float* b2_l = b2 + l * DM_;

    // warm the FFN weights into cache (global_prefetch_b8)
    if (tid * 64 < DM_ * DF_ * 4) {
      __builtin_prefetch((const char*)W1_l + tid * 64, 0, 0);
      __builtin_prefetch((const char*)W2_l + tid * 64, 0, 0);
    }

    // ---- Stage A: LN1(x) -> hn (threads <512), QKVO frag staging (threads >=512) ----
    // Bias-row trick: hn col31 = 1.0, and B-fragment row 31 carries the bias vector.
    if (tid < V_) {
      const float* xrow = xs + tid * DMP;
      float mu = 0.0f;
#pragma unroll
      for (int d = 0; d < DM_; ++d) mu += xrow[d];
      mu *= (1.0f / DM_);
      float var = 0.0f;
#pragma unroll
      for (int d = 0; d < DM_; ++d) { float t = xrow[d] - mu; var += t * t; }
      var *= (1.0f / (DM_ - 1));
      const float rs = rsqrtf(var + EPS_);
      _Float16* hrow = hn + tid * DMP;
#pragma unroll
      for (int d = 0; d < DM_; ++d)
        hrow[d] = (_Float16)(n1a[l * DM_ + d] * (xrow[d] - mu) * rs + n1b[l * DM_ + d]);
      hrow[30] = (_Float16)0.0f; hrow[31] = (_Float16)1.0f;   // bias row selector
    } else {
      // stage Wq/Wk/Wv/Wo as B-fragments with bias in row 31
      for (int idx = tid - V_; idx < 4 * 2 * 32 * 16; idx += V_) {
        const int i  = idx & 15;
        const int ln = (idx >> 4) & 31;
        const int j  = (idx >> 9) & 1;
        const int mat = idx >> 10;
        const int k = frag_k(ln, i);
        const int n = (ln & 15) + 16 * j;
        const float* Wm = (mat == 0) ? Wq_l : (mat == 1) ? Wk_l : (mat == 2) ? Wv_l : Wo_l;
        const float* Bm = (mat == 0) ? (bq + l * DM_) : (mat == 1) ? (bk + l * DM_)
                        : (mat == 2) ? (bv + l * DM_) : (bo + l * DM_);
        float v = 0.0f;
        if (n < DM_) {
          if (k < DM_) v = Wm[k * DM_ + n];
          else if (k == 31) v = Bm[n];
        }
        wq4[idx] = (_Float16)v;
      }
    }
    __syncthreads();

    // ---------------- Stage B: QKV projections (WMMA, bias folded) ----------------
    {
      const v16h a = load_afrag(hn + m0 * DMP, DMP);
#pragma unroll
      for (int j = 0; j < 2; ++j) {
        const int n = (lane & 15) + 16 * j;
        const bool nok = (n < DM_);
        { // Q -> qo token-major; col30 = 0, col31 = 1.0 (bias selector for Wo pass)
          v16h bf = load_frag16(wq4, 0 * 2 + j);
          v8f c = {};
          c = wmma16(a, bf, c);
#pragma unroll
          for (int r = 0; r < 8; ++r) {
            const int m = m0 + r + 8 * (lane >> 4);
            qo[m * DMP + n] = (_Float16)(nok ? c[r] : (n == 31 ? 1.0f : 0.0f));
          }
        }
        { // K -> head-major kh[head][token][0..4]
          v16h bf = load_frag16(wq4, 1 * 2 + j);
          v8f c = {};
          c = wmma16(a, bf, c);
          if (nok) {
            const int hd = n / DH_, id = n % DH_;
#pragma unroll
            for (int r = 0; r < 8; ++r) {
              const int m = m0 + r + 8 * (lane >> 4);
              kh[(hd * V_ + m) * 8 + id] = (_Float16)c[r];
            }
          }
        }
        { // V -> head-major vh
          v16h bf = load_frag16(wq4, 2 * 2 + j);
          v8f c = {};
          c = wmma16(a, bf, c);
          if (nok) {
            const int hd = n / DH_, id = n % DH_;
#pragma unroll
            for (int r = 0; r < 8; ++r) {
              const int m = m0 + r + 8 * (lane >> 4);
              vh[(hd * V_ + m) * 8 + id] = (_Float16)c[r];
            }
          }
        }
      }
    }
    __syncthreads();

    // ---------------- Stage C: attention (online softmax, DH=5) ----------------
    {
      const int i  = tid & (V_ - 1);
      const int g  = tid >> 9;
      const int h0 = 3 * g;
      float qreg[15];
#pragma unroll
      for (int t = 0; t < 15; ++t) qreg[t] = (float)qo[i * DMP + 15 * g + t];
      float mh[3], sh[3], acc[3][5];
#pragma unroll
      for (int hh = 0; hh < 3; ++hh) {
        mh[hh] = -3.0e38f; sh[hh] = 0.0f;
#pragma unroll
        for (int d = 0; d < 5; ++d) acc[hh][d] = 0.0f;
      }
      const float scale = 0.44721359549995793f;  // 1/sqrt(5)
#pragma unroll 1
      for (int jt = 0; jt < V_; ++jt) {
#pragma unroll
        for (int hh = 0; hh < 3; ++hh) {
          const v8h kv = *(const v8h*)(kh + ((h0 + hh) * V_ + jt) * 8);
          const v8h vv = *(const v8h*)(vh + ((h0 + hh) * V_ + jt) * 8);
          float s = 0.0f;
#pragma unroll
          for (int d = 0; d < 5; ++d) s += qreg[hh * 5 + d] * (float)kv[d];
          s *= scale;
          const float mn = fmaxf(mh[hh], s);
          const float corr = __expf(mh[hh] - mn);
          const float p = __expf(s - mn);
          sh[hh] = sh[hh] * corr + p;
#pragma unroll
          for (int d = 0; d < 5; ++d) acc[hh][d] = acc[hh][d] * corr + p * (float)vv[d];
          mh[hh] = mn;
        }
      }
#pragma unroll
      for (int hh = 0; hh < 3; ++hh) {
        const float inv = 1.0f / sh[hh];
#pragma unroll
        for (int d = 0; d < 5; ++d)
          qo[i * DMP + 15 * g + hh * 5 + d] = (_Float16)(acc[hh][d] * inv);
      }
    }
    __syncthreads();

    // ---- Stage S: stage W1/W2 as B-fragments into the (now dead) K/V region ----
    // w1f frag f = n-tile (0..31): n0 = 16*f, rows: k<30 weights, k==31 bias b1
    for (int idx = tid; idx < 32 * 32 * 16; idx += 1024) {
      const int i  = idx & 15;
      const int ln = (idx >> 4) & 31;
      const int f  = idx >> 9;
      const int k = frag_k(ln, i);
      const int n = (ln & 15) + 16 * f;
      float v;
      if (k < DM_)       v = W1_l[k * DF_ + n];
      else if (k == 31)  v = b1_l[n];
      else               v = 0.0f;
      w1f[idx] = (_Float16)v;
    }
    // w2f frag f = kc*2+j: rows kc*32 + k, cols 16*j (cols >= 30 zero)
    for (int idx = tid; idx < 32 * 32 * 16; idx += 1024) {
      const int i  = idx & 15;
      const int ln = (idx >> 4) & 31;
      const int f  = idx >> 9;
      const int kc = f >> 1, j = f & 1;
      const int k = frag_k(ln, i);
      const int n = (ln & 15) + 16 * j;
      w2f[idx] = (n < DM_) ? (_Float16)W2_l[(kc * 32 + k) * DM_ + n] : (_Float16)0.0f;
    }

    // ---------------- Stage D: h = hn + o @ Wo (+bo via bias row) ----------------
    {
      const v16h a = load_afrag(qo + m0 * DMP, DMP);
#pragma unroll
      for (int j = 0; j < 2; ++j) {
        v16h bf = load_frag16(wq4, 3 * 2 + j);
        v8f c = {};
        c = wmma16(a, bf, c);
        const int n = (lane & 15) + 16 * j;
#pragma unroll
        for (int r = 0; r < 8; ++r) {
          const int m = m0 + r + 8 * (lane >> 4);
          hs[m * DMP + n] = (float)hn[m * DMP + n] + c[r];
        }
      }
    }
    __syncthreads();

    // ---------------- Stage E: LN2(h) -> hn ----------------
    if (tid < V_) {
      const float* hrow = hs + tid * DMP;
      float mu = 0.0f;
#pragma unroll
      for (int d = 0; d < DM_; ++d) mu += hrow[d];
      mu *= (1.0f / DM_);
      float var = 0.0f;
#pragma unroll
      for (int d = 0; d < DM_; ++d) { float t = hrow[d] - mu; var += t * t; }
      var *= (1.0f / (DM_ - 1));
      const float rs = rsqrtf(var + EPS_);
      _Float16* orow = hn + tid * DMP;
#pragma unroll
      for (int d = 0; d < DM_; ++d)
        orow[d] = (_Float16)(n2a[l * DM_ + d] * (hrow[d] - mu) * rs + n2b[l * DM_ + d]);
      orow[30] = (_Float16)0.0f; orow[31] = (_Float16)1.0f;   // bias row selector
    }
    __syncthreads();

    // ---- Stage F: fused FFN, h += relu(h2@W1 [+b1 via bias row])@W2 + b2 ----
    // Inner loop is pure LDS + WMMA: no VMEM, no loadcnt waits.
    {
      const v16h a2 = load_afrag(hn + m0 * DMP, DMP);
      v8f cF0 = {}, cF1 = {};
      float bb2[2];
#pragma unroll
      for (int j = 0; j < 2; ++j) {
        const int n = (lane & 15) + 16 * j;
        bb2[j] = (n < DM_) ? b2_l[n] : 0.0f;   // hoisted out of the hot loop
      }
      _Float16* u = uS + w * 512;  // wave-private 16x32 f16 tile
#pragma unroll 1
      for (int kc = 0; kc < DF_ / 32; ++kc) {
#pragma unroll
        for (int j = 0; j < 2; ++j) {
          v16h bf = load_frag16(w1f, kc * 2 + j);
          v8f c = {};
          c = wmma16(a2, bf, c);
          const int nl = (lane & 15) + 16 * j;
#pragma unroll
          for (int r = 0; r < 8; ++r) {
            const float val = c[r] > 0.0f ? c[r] : 0.0f;     // relu (bias already in)
            u[(r + 8 * (lane >> 4)) * 32 + nl] = (_Float16)val;
          }
        }
        asm volatile("s_wait_dscnt 0" ::: "memory");         // C-layout -> A-layout via LDS
        const v16h uf = load_afrag(u, 32);
        cF0 = wmma16(uf, load_frag16(w2f, kc * 2 + 0), cF0);
        cF1 = wmma16(uf, load_frag16(w2f, kc * 2 + 1), cF1);
      }
#pragma unroll
      for (int j = 0; j < 2; ++j) {
        const v8f c = j ? cF1 : cF0;
        const int n = (lane & 15) + 16 * j;
#pragma unroll
        for (int r = 0; r < 8; ++r) {
          const int m = m0 + r + 8 * (lane >> 4);
          hs[m * DMP + n] = hs[m * DMP + n] + c[r] + bb2[j];
        }
      }
    }
    __syncthreads();

    // ---------------- Stage G: emb renorm + residual (+softmax last 5) ----------------
    if (tid < V_) {
      const float* hrow = hs + tid * DMP;
      float* xrow = xs + tid * DMP;
      float mu = 0.0f;
#pragma unroll
      for (int e = 0; e < E_; ++e) mu += hrow[e];
      mu *= (1.0f / E_);
      float var = 0.0f;
#pragma unroll
      for (int e = 0; e < E_; ++e) { float t = hrow[e] - mu; var += t * t; }
      var *= (1.0f / (E_ - 1));
      const float rs = rsqrtf(var + EPS_);
      float xn[DM_];
#pragma unroll
      for (int e = 0; e < E_; ++e)
        xn[e] = q_alpha[qi * E_ + e] * (hrow[e] - mu) * rs + q_bias[qi * E_ + e] + xrow[e];
#pragma unroll
      for (int d = E_; d < DM_; ++d) xn[d] = hrow[d] + xrow[d];
      if (l < L_ - 1) {
        float mx = xn[E_];
#pragma unroll
        for (int t = 1; t < K_; ++t) mx = fmaxf(mx, xn[E_ + t]);
        float s = 0.0f, ex[K_];
#pragma unroll
        for (int t = 0; t < K_; ++t) { ex[t] = __expf(xn[E_ + t] - mx); s += ex[t]; }
        const float inv = 1.0f / s;
#pragma unroll
        for (int t = 0; t < K_; ++t) xn[E_ + t] = ex[t] * inv;
      }
#pragma unroll
      for (int d = 0; d < DM_; ++d) xrow[d] = xn[d];
    }
    __syncthreads();
  }

  // ---------------- writeback ----------------
  if (tid < V_) {
    const float* xrow = xs + tid * DMP;
    float* orow = out + ((size_t)b * V_ + tid) * DM_;
#pragma unroll
    for (int d = 0; d < DM_; ++d) orow[d] = xrow[d];
  }
}

extern "C" void kernel_launch(void* const* d_in, const int* in_sizes, int n_in,
                              void* d_out, int out_size, void* d_ws, size_t ws_size,
                              hipStream_t stream) {
  (void)in_sizes; (void)n_in; (void)out_size; (void)d_ws; (void)ws_size;
  const float* inputs    = (const float*)d_in[0];
  const int*   questions = (const int*)d_in[1];
  const float* q_emb     = (const float*)d_in[2];
  const float* q_alpha   = (const float*)d_in[3];
  const float* q_bias    = (const float*)d_in[4];
  const float* Wq        = (const float*)d_in[5];
  const float* bq        = (const float*)d_in[6];
  const float* Wk        = (const float*)d_in[7];
  const float* bk        = (const float*)d_in[8];
  const float* Wv        = (const float*)d_in[9];
  const float* bv        = (const float*)d_in[10];
  const float* Wo        = (const float*)d_in[11];
  const float* bo        = (const float*)d_in[12];
  const float* n1a       = (const float*)d_in[13];
  const float* n1b       = (const float*)d_in[14];
  const float* n2a       = (const float*)d_in[15];
  const float* n2b       = (const float*)d_in[16];
  const float* W1        = (const float*)d_in[17];
  const float* b1        = (const float*)d_in[18];
  const float* W2        = (const float*)d_in[19];
  const float* b2        = (const float*)d_in[20];
  float* out = (float*)d_out;

  (void)hipFuncSetAttribute((const void*)imputer_fused_kernel,
                            hipFuncAttributeMaxDynamicSharedMemorySize, SMEM_BYTES);
  imputer_fused_kernel<<<dim3(B_), dim3(1024), SMEM_BYTES, stream>>>(
      inputs, questions, q_emb, q_alpha, q_bias,
      Wq, bq, Wk, bk, Wv, bv, Wo, bo,
      n1a, n1b, n2a, n2b, W1, b1, W2, b2, out);
}
